// KcatPrediction_69063074119826
// MI455X (gfx1250) — compile-verified, matrix operand order checked by reading, not run on hardware
//
#include <hip/hip_runtime.h>
#include <math.h>

// ---------------------------------------------------------------------------
// Problem constants
// ---------------------------------------------------------------------------
#define DIM     128
#define NA      4096
#define NWORDS  16384
#define KHALF   11
#define KSIZE   23          // 2*KHALF+1
#define KTAPS   (KSIZE*KSIZE)
#define KT_ADJ  (NA / 32)   // 128 k-tiles for the adjacency GEMM

typedef __attribute__((ext_vector_type(16))) __bf16 v16bf;
typedef __attribute__((ext_vector_type(8)))  float  v8f;

#if defined(__AMDGCN__) && __has_builtin(__builtin_amdgcn_global_load_async_to_lds_b32)
#define USE_ASYNC_LDS 1
#else
#define USE_ASYNC_LDS 0
#endif

#if USE_ASYNC_LDS
// async global->LDS b32 copy (tracked on ASYNCcnt). Builtin takes int* args.
__device__ __forceinline__ void async_copy_b32(const float* g, float* l) {
    __builtin_amdgcn_global_load_async_to_lds_b32(
        (__attribute__((address_space(1))) int*)(uintptr_t)g,
        (__attribute__((address_space(3))) int*)
            (__attribute__((address_space(3))) void*)l,
        0, 0);
}
#endif

// ---------------------------------------------------------------------------
// 1) Row gather: out[r,:] = emb[idx[r],:]
// ---------------------------------------------------------------------------
__global__ void gather_rows_kernel(const float* __restrict__ emb,
                                   const int* __restrict__ idx,
                                   float* __restrict__ out, int nrows) {
    int i = blockIdx.x * 256 + threadIdx.x;
    int r = i >> 7;
    int d = i & 127;
    if (r < nrows) out[i] = emb[(size_t)idx[r] * DIM + d];
}

// ---------------------------------------------------------------------------
// 2) One-time pack: adjacency f32 {0,1} -> bf16 (lossless) in WMMA A-fragment
//    lane order. A-frag layout (16x32, wave32): lane = (m&15) + 16*half,
//    element e -> K = (e<8?0:16) + half*8 + (e&7).
//    Each thread converts one 8-float chunk -> one contiguous 16B store.
// ---------------------------------------------------------------------------
__global__ void pack_adj_kernel(const float* __restrict__ A,
                                __bf16* __restrict__ Apk) {
    const size_t id = (size_t)blockIdx.x * 256 + threadIdx.x; // NA*NA/8 threads
    const int m  = (int)(id >> 9);         // 512 chunks of 8 per row
    const int ch = (int)(id & 511);
    const int k0 = ch * 8;
    const int ktile = k0 >> 5;
    const int kk    = k0 & 31;
    const int group = (kk >> 4) & 1;
    const int half  = (kk >> 3) & 1;
    const int mtile = m >> 4;
    const int lane  = (m & 15) | (half << 4);
    const int e0    = group * 8;

    const float4* src = (const float4*)(A + (size_t)m * NA + k0);
    const float4 a0 = src[0], a1 = src[1];
    union { __bf16 e[8]; uint4 u; } o;
    o.e[0] = (__bf16)a0.x; o.e[1] = (__bf16)a0.y;
    o.e[2] = (__bf16)a0.z; o.e[3] = (__bf16)a0.w;
    o.e[4] = (__bf16)a1.x; o.e[5] = (__bf16)a1.y;
    o.e[6] = (__bf16)a1.z; o.e[7] = (__bf16)a1.w;

    __bf16* dst = Apk + (((size_t)(mtile * KT_ADJ + ktile)) * 32 + lane) * 16 + e0;
    *(uint4*)dst = o.u;
}

// ---------------------------------------------------------------------------
// 3) Linear + ReLU: y = relu(x @ W^T + b), x:[rows,128], W:[128,128]
//    Optionally also emits y split as bf16 hi/lo in WMMA-B fragment order.
// ---------------------------------------------------------------------------
__global__ void linear_relu_kernel(const float* __restrict__ x,
                                   const float* __restrict__ W,
                                   const float* __restrict__ b,
                                   float* __restrict__ y,
                                   __bf16* __restrict__ pk_hi,
                                   __bf16* __restrict__ pk_lo,
                                   int rows, int write_packed) {
    __shared__ float sx[16 * DIM];
    const int t = threadIdx.x;                 // 256 threads
    const int rbase = blockIdx.x * 16;
    for (int i = t; i < 16 * DIM; i += 256)
        sx[i] = x[(size_t)rbase * DIM + i];
    __syncthreads();

    const int c  = t & 127;
    const int rg = t >> 7;                     // 0/1 -> rows rg*8 .. rg*8+7
    float acc[8];
#pragma unroll
    for (int i = 0; i < 8; ++i) acc[i] = 0.f;

    const float* wr = W + (size_t)c * DIM;
    for (int k = 0; k < DIM; ++k) {
        const float wv = wr[k];
#pragma unroll
        for (int i = 0; i < 8; ++i)
            acc[i] += sx[(rg * 8 + i) * DIM + k] * wv;
    }

    const float bias = b[c];
    const int KT = rows >> 5;                  // k-tiles of 32 for packed layout
#pragma unroll
    for (int i = 0; i < 8; ++i) {
        const int row = rbase + rg * 8 + i;
        float v = acc[i] + bias;
        v = fmaxf(v, 0.f);
        y[(size_t)row * DIM + c] = v;
        if (write_packed) {
            const __bf16 hi = (__bf16)v;
            const __bf16 lo = (__bf16)(v - (float)hi);
            const int kk    = row & 31;
            const int ktile = row >> 5;
            const int group = kk >> 4;
            const int rem   = kk & 15;
            const int half  = rem >> 3;
            const int e     = group * 8 + (rem & 7);
            const int lane  = (c & 15) + (half << 4);
            const int ntile = c >> 4;
            const size_t off = (((size_t)(ntile * KT + ktile)) * 32 + lane) * 16 + e;
            pk_hi[off] = hi;
            pk_lo[off] = lo;
        }
    }
}

// ---------------------------------------------------------------------------
// 4) xs += A @ hs via WMMA bf16. A pre-packed (exact), hs split hi+lo for
//    ~fp32 accuracy. All operands loaded in fragment order: 6 b128 loads +
//    2 v_wmma per k-step, no conversions in the hot loop.
// ---------------------------------------------------------------------------
__device__ inline v16bf load_bfrag(const __bf16* p) {
    union { uint4 u[2]; v16bf v; } t;
    const uint4* q = (const uint4*)p;
    t.u[0] = q[0];
    t.u[1] = q[1];
    return t.v;
}

__global__ void adj_gemm_kernel(const __bf16* __restrict__ Apk,
                                const __bf16* __restrict__ Bhi,
                                const __bf16* __restrict__ Blo,
                                float* __restrict__ xs) {       // [4096,128]
    const int mtile = blockIdx.x;              // 0..255
    const int wave  = threadIdx.x >> 5;        // ntile 0..7
    const int lane  = threadIdx.x & 31;
    const int half  = lane >> 4;

    v8f acc = {0.f, 0.f, 0.f, 0.f, 0.f, 0.f, 0.f, 0.f};
    const __bf16* ap = Apk + (((size_t)mtile * KT_ADJ) * 32 + lane) * 16;
    const __bf16* bh = Bhi + (((size_t)wave  * KT_ADJ) * 32 + lane) * 16;
    const __bf16* bl = Blo + (((size_t)wave  * KT_ADJ) * 32 + lane) * 16;

    for (int ks = 0; ks < KT_ADJ; ++ks) {
        const v16bf af = load_bfrag(ap);
        const v16bf fh = load_bfrag(bh);
        const v16bf fl = load_bfrag(bl);
        __builtin_prefetch(ap + 1024, 0, 3);   // two k-tiles ahead
        acc = __builtin_amdgcn_wmma_f32_16x16x32_bf16(
                  false, af, false, fh, (short)0, acc, false, false);
        acc = __builtin_amdgcn_wmma_f32_16x16x32_bf16(
                  false, af, false, fl, (short)0, acc, false, false);
        ap += 512;
        bh += 512;
        bl += 512;
    }

    // C layout: VGPR r, lane -> row = r + 8*half, col = lane&15 (within tile)
    union { float e[8]; v8f v; } cu;
    cu.v = acc;
    const int col = wave * 16 + (lane & 15);
#pragma unroll
    for (int r = 0; r < 8; ++r) {
        const int row = mtile * 16 + r + 8 * half;
        xs[(size_t)row * DIM + col] += cu.e[r];
    }
}

// ---------------------------------------------------------------------------
// 5) compound = mean(xs, rows); h = relu(W_att @ compound + b); zero pacc
// ---------------------------------------------------------------------------
__global__ void compound_h_kernel(const float* __restrict__ xs,
                                  const float* __restrict__ Watt,
                                  const float* __restrict__ batt,
                                  float* __restrict__ compound,
                                  float* __restrict__ hvec,
                                  float* __restrict__ pacc) {
    __shared__ float sc[DIM];
    const int t = threadIdx.x;                 // 128 threads
    float s = 0.f;
    for (int r = 0; r < NA; ++r) s += xs[(size_t)r * DIM + t];
    s *= (1.f / (float)NA);
    sc[t] = s;
    compound[t] = s;
    pacc[t] = 0.f;                             // reset protein accumulator
    __syncthreads();
    float acc = batt[t];
    const float* wr = Watt + (size_t)t * DIM;
    for (int k = 0; k < DIM; ++k) acc += wr[k] * sc[k];
    hvec[t] = fmaxf(acc, 0.f);
}

// ---------------------------------------------------------------------------
// 6) 23x23 SAME conv + bias + leaky_relu(0.01) over [16384,128].
//    Tile staged into LDS via CDNA5 async global->LDS copies (ASYNCcnt),
//    with plain-load fallback if the builtin is unavailable.
// ---------------------------------------------------------------------------
#define CROWS 32
#define LDW   152   // 128 + 22 halo, padded to 152
__global__ void conv23_kernel(const float* __restrict__ in,
                              const float* __restrict__ w,   // [23*23]
                              const float* __restrict__ bias,
                              float* __restrict__ out) {
    __shared__ float tile[(CROWS + 2 * KHALF) * LDW];
    __shared__ float wl[KTAPS];
    const int t = threadIdx.x;                 // 256 threads
    const int rbase = blockIdx.x * CROWS;

    for (int i = t; i < KTAPS; i += 256) wl[i] = w[i];
    for (int i = t; i < (CROWS + 2 * KHALF) * LDW; i += 256) {
        const int lr = i / LDW, lc = i % LDW;
        const int gr = rbase - KHALF + lr;
        const int gc = lc - KHALF;
        const bool inr = (gr >= 0 && gr < NWORDS && gc >= 0 && gc < DIM);
#if USE_ASYNC_LDS
        if (inr) {
            async_copy_b32(in + (size_t)gr * DIM + gc, tile + i);
        } else {
            tile[i] = 0.f;
        }
#else
        tile[i] = inr ? in[(size_t)gr * DIM + gc] : 0.f;
#endif
    }
#if USE_ASYNC_LDS
    asm volatile("s_wait_asynccnt 0" ::: "memory");
#endif
    __syncthreads();

    const float bb = bias[0];
    for (int o = 0; o < (CROWS * DIM) / 256; ++o) {
        const int idx = t + o * 256;
        const int lr = idx >> 7;
        const int c  = idx & 127;
        float acc = bb;
        for (int ky = 0; ky < KSIZE; ++ky) {
            const float* trow = &tile[(lr + ky) * LDW + c];
            const float* wrow = &wl[ky * KSIZE];
#pragma unroll
            for (int kx = 0; kx < KSIZE; ++kx)
                acc += trow[kx] * wrow[kx];
        }
        acc = (acc > 0.f) ? acc : 0.01f * acc;
        out[(size_t)(rbase + lr) * DIM + c] = acc;
    }
}

// ---------------------------------------------------------------------------
// 7) attention: w_i = tanh(dot(h, hs_i)); pacc += sum_i w_i * hs_i
// ---------------------------------------------------------------------------
__global__ void attention_kernel(const float* __restrict__ hs,   // [16384,128]
                                 const float* __restrict__ hvec, // [128]
                                 float* __restrict__ pacc) {     // [128]
    __shared__ float sp[DIM];
    const int t = threadIdx.x;                 // 256 = 8 wave32s
    if (t < DIM) sp[t] = 0.f;
    __syncthreads();
    const int wave = t >> 5;
    const int lane = t & 31;
    const int rbase = blockIdx.x * 32 + wave * 4;
    for (int rr = 0; rr < 4; ++rr) {
        const float* hr = hs + (size_t)(rbase + rr) * DIM;
        float d = 0.f;
#pragma unroll
        for (int i = 0; i < 4; ++i) d += hr[lane + 32 * i] * hvec[lane + 32 * i];
        for (int off = 16; off > 0; off >>= 1) d += __shfl_xor(d, off, 32);
        const float wgt = tanhf(d);
#pragma unroll
        for (int i = 0; i < 4; ++i) {
            const int k = lane + 32 * i;
            atomicAdd(&sp[k], wgt * hr[k]);    // ds_add_f32
        }
    }
    __syncthreads();
    if (t < DIM) atomicAdd(&pacc[t], sp[t]);   // global_atomic_add_f32
}

// ---------------------------------------------------------------------------
// 8) output MLP head (256-wide, 2 relu layers + scalar)
// ---------------------------------------------------------------------------
__global__ void out_mlp_kernel(const float* __restrict__ compound,
                               const float* __restrict__ pacc,
                               const float* __restrict__ Wout,  // [2,256,256]
                               const float* __restrict__ bout,  // [2,256]
                               const float* __restrict__ Wint,  // [256]
                               const float* __restrict__ bint,  // [1]
                               float* __restrict__ out) {
    __shared__ float cat[2 * DIM];
    __shared__ float nxt[2 * DIM];
    const int t = threadIdx.x;                 // 256 threads
    cat[t] = (t < DIM) ? compound[t] : pacc[t - DIM] * (1.f / (float)NWORDS);
    __syncthreads();
    for (int l = 0; l < 2; ++l) {
        const float* w = Wout + (size_t)l * 256 * 256 + (size_t)t * 256;
        float acc = bout[l * 256 + t];
        for (int k = 0; k < 256; ++k) acc += w[k] * cat[k];
        nxt[t] = fmaxf(acc, 0.f);
        __syncthreads();
        cat[t] = nxt[t];
        __syncthreads();
    }
    if (t == 0) {
        float acc = bint[0];
        for (int k = 0; k < 256; ++k) acc += Wint[k] * cat[k];
        out[0] = acc;
    }
}

// ---------------------------------------------------------------------------
// Host launcher
// ---------------------------------------------------------------------------
extern "C" void kernel_launch(void* const* d_in, const int* in_sizes, int n_in,
                              void* d_out, int out_size, void* d_ws, size_t ws_size,
                              hipStream_t stream) {
    const int*   fingerprints = (const int*)  d_in[0];
    const float* adjacency    = (const float*)d_in[1];
    const int*   words        = (const int*)  d_in[2];
    const float* emb_fp       = (const float*)d_in[3];
    const float* emb_word     = (const float*)d_in[4];
    const float* W_gnn_w      = (const float*)d_in[5];
    const float* W_gnn_b      = (const float*)d_in[6];
    const float* conv_w       = (const float*)d_in[7];
    const float* conv_b       = (const float*)d_in[8];
    const float* W_att_w      = (const float*)d_in[9];
    const float* W_att_b      = (const float*)d_in[10];
    const float* W_out_w      = (const float*)d_in[11];
    const float* W_out_b      = (const float*)d_in[12];
    const float* W_int_w      = (const float*)d_in[13];
    const float* W_int_b      = (const float*)d_in[14];
    float* out = (float*)d_out;

    // workspace layout (bytes)
    char* w = (char*)d_ws;
    float*  xs    = (float*) (w);                           // 2 MB
    float*  hs    = (float*) (w + (size_t)2  * (1u << 20)); // 2 MB
    __bf16* pkhi  = (__bf16*)(w + (size_t)4  * (1u << 20)); // 1 MB
    __bf16* pklo  = (__bf16*)(w + (size_t)5  * (1u << 20)); // 1 MB
    float*  img0  = (float*) (w + (size_t)6  * (1u << 20)); // 8 MB
    float*  img1  = (float*) (w + (size_t)14 * (1u << 20)); // 8 MB
    float*  small = (float*) (w + (size_t)22 * (1u << 20)); // few KB
    __bf16* Apk   = (__bf16*)(w + (size_t)24 * (1u << 20)); // 32 MB
    float* compound = small;
    float* hvec     = small + 128;
    float* pacc     = small + 256;

    // ---- one-time adjacency pack (reused by all 3 GNN layers) ----
    pack_adj_kernel<<<(NA * NA / 8) / 256, 256, 0, stream>>>(adjacency, Apk);

    // ---- compound GNN ----
    gather_rows_kernel<<<(NA * DIM) / 256, 256, 0, stream>>>(emb_fp, fingerprints, xs, NA);
    for (int l = 0; l < 3; ++l) {
        linear_relu_kernel<<<NA / 16, 256, 0, stream>>>(
            xs, W_gnn_w + (size_t)l * DIM * DIM, W_gnn_b + (size_t)l * DIM,
            hs, pkhi, pklo, NA, 1);
        adj_gemm_kernel<<<NA / 16, 256, 0, stream>>>(Apk, pkhi, pklo, xs);
    }
    compound_h_kernel<<<1, 128, 0, stream>>>(xs, W_att_w, W_att_b, compound, hvec, pacc);

    // ---- protein CNN ----
    gather_rows_kernel<<<(NWORDS * DIM) / 256, 256, 0, stream>>>(emb_word, words, img0, NWORDS);
    float* ia = img0;
    float* ib = img1;
    for (int l = 0; l < 3; ++l) {
        conv23_kernel<<<NWORDS / CROWS, 256, 0, stream>>>(
            ia, conv_w + (size_t)l * KTAPS, conv_b + l, ib);
        float* tmp = ia; ia = ib; ib = tmp;
    }
    // final conv result is in `ia`; write attention features into `ib`
    float* hs_att = ib;
    linear_relu_kernel<<<NWORDS / 16, 256, 0, stream>>>(
        ia, W_att_w, W_att_b, hs_att, nullptr, nullptr, NWORDS, 0);
    attention_kernel<<<NWORDS / 32, 256, 0, stream>>>(hs_att, hvec, pacc);

    // ---- output head ----
    out_mlp_kernel<<<1, 256, 0, stream>>>(compound, pacc, W_out_w, W_out_b,
                                          W_int_w, W_int_b, out);
    (void)in_sizes; (void)n_in; (void)out_size; (void)ws_size;
}
